// EdgeOnlyConv_19662360281539
// MI455X (gfx1250) — matrix operand order repaired and checked by reading, not compiled, not run on hardware
//
#include <hip/hip_runtime.h>

#define N_IN_NODE 128
#define N_IN_EDGE 64
#define N_OUT     128
#define K_TOT     (2 * N_IN_NODE + N_IN_EDGE)   // 320
#define M_TILE    32                             // edges per block
#define LDS_STRIDE 328                           // padded (ushort units) to spread LDS banks

typedef __attribute__((ext_vector_type(16))) _Float16      v16h;
typedef __attribute__((ext_vector_type(8)))  float         v8f;
typedef __attribute__((ext_vector_type(4)))  float         v4f;
typedef __attribute__((ext_vector_type(16))) unsigned short v16us;
typedef __attribute__((ext_vector_type(8)))  unsigned short v8us;
typedef __attribute__((ext_vector_type(4)))  unsigned int  v4u;

// two f32 -> packed f16 pair in one v_cvt_pk_rtz_f16_f32
__device__ __forceinline__ unsigned int pkh(float lo, float hi) {
    return __builtin_bit_cast(unsigned int, __builtin_amdgcn_cvt_pkrtz(lo, hi));
}

// CDNA5 async copy: 16 bytes global -> LDS, tracked with ASYNCcnt (no VGPR data path)
__device__ __forceinline__ void async_g2l_b128(unsigned int lds_byte_off, const void* gaddr) {
    asm volatile("global_load_async_to_lds_b128 %0, %1, off"
                 :: "v"(lds_byte_off), "v"(gaddr)
                 : "memory");
}
__device__ __forceinline__ void wait_async0() {
    asm volatile("s_wait_asynccnt 0x0" ::: "memory");
}

// ---------- one-shot: f32 array -> f16 bits (n must be a multiple of 8) ----------
__global__ void cvt_f32_f16(const float* __restrict__ src,
                            unsigned short* __restrict__ dst, int n) {
    int i = (blockIdx.x * blockDim.x + threadIdx.x) * 8;
    if (i + 7 < n) {
        const v4f* p = (const v4f*)(src + i);
        v4f a = p[0], b = p[1];
        v4u q = { pkh(a.x, a.y), pkh(a.z, a.w), pkh(b.x, b.y), pkh(b.z, b.w) };
        *(v4u*)(dst + i) = q;
    }
}

// ---------- main kernel: 32 edges x 128 outputs per block ----------
// MODE: 0 = no workspace, 1 = W preconverted, 2 = W and x preconverted (async staging)
template <int MODE>
__global__ __launch_bounds__(256, 2)
void edgeconv_wmma_f16(const float* __restrict__ x,
                       const unsigned short* __restrict__ xh,
                       const int*   __restrict__ ei,
                       const float* __restrict__ ea,
                       const float* __restrict__ W,
                       const unsigned short* __restrict__ Wh,
                       const float* __restrict__ bias,
                       float*       __restrict__ out,
                       int n_edges) {
    __shared__ unsigned short sfeat[M_TILE * LDS_STRIDE];

    const int tid   = threadIdx.x;
    const int edge0 = blockIdx.x * M_TILE;

    // ---- stage: gather 32x320 panel into f16 LDS (8 threads / edge) ----
    {
        const int e = tid >> 3;
        const int j = tid & 7;
        int ge = edge0 + e;
        if (ge >= n_edges) ge = n_edges - 1;            // clamp tail; stores guarded later
        const int src = ei[ge];
        const int dst = ei[n_edges + ge];
        unsigned short* srow = &sfeat[e * LDS_STRIDE];

        if (MODE == 2) {
            // x already f16: pure async DMA global->LDS, 32 B per thread per row
            const unsigned int srow_b = (unsigned int)(e * (LDS_STRIDE * 2)); // LDS byte offset
            const unsigned short* xs = xh + (size_t)src * N_IN_NODE + j * 16;
            const unsigned short* xd = xh + (size_t)dst * N_IN_NODE + j * 16;
            async_g2l_b128(srow_b + j * 32,            xs);
            async_g2l_b128(srow_b + j * 32 + 16,       xs + 8);
            async_g2l_b128(srow_b + 256 + j * 32,      xd);
            async_g2l_b128(srow_b + 256 + j * 32 + 16, xd + 8);
        } else {
            // x[src]: 16 floats per thread
            const v4f* p = (const v4f*)(x + (size_t)src * N_IN_NODE + j * 16);
            v4f a0 = p[0], a1 = p[1], a2 = p[2], a3 = p[3];
            v4u q0 = { pkh(a0.x, a0.y), pkh(a0.z, a0.w), pkh(a1.x, a1.y), pkh(a1.z, a1.w) };
            v4u q1 = { pkh(a2.x, a2.y), pkh(a2.z, a2.w), pkh(a3.x, a3.y), pkh(a3.z, a3.w) };
            *(v4u*)(srow + j * 16)     = q0;
            *(v4u*)(srow + j * 16 + 8) = q1;
            // x[dest]: 16 floats per thread
            const v4f* pd = (const v4f*)(x + (size_t)dst * N_IN_NODE + j * 16);
            v4f b0 = pd[0], b1 = pd[1], b2 = pd[2], b3 = pd[3];
            v4u r0 = { pkh(b0.x, b0.y), pkh(b0.z, b0.w), pkh(b1.x, b1.y), pkh(b1.z, b1.w) };
            v4u r1 = { pkh(b2.x, b2.y), pkh(b2.z, b2.w), pkh(b3.x, b3.y), pkh(b3.z, b3.w) };
            *(v4u*)(srow + N_IN_NODE + j * 16)     = r0;
            *(v4u*)(srow + N_IN_NODE + j * 16 + 8) = r1;
        }

        // edge_attr: 8 floats per thread (read-once stream -> NT), convert in-register
        {
            const v4f* p = (const v4f*)(ea + (size_t)ge * N_IN_EDGE + j * 8);
            v4f c0 = __builtin_nontemporal_load(p);
            v4f c1 = __builtin_nontemporal_load(p + 1);
            v4u q = { pkh(c0.x, c0.y), pkh(c0.z, c0.w), pkh(c1.x, c1.y), pkh(c1.z, c1.w) };
            *(v4u*)(srow + 2 * N_IN_NODE + j * 8) = q;
        }
    }
    if (MODE == 2) wait_async0();     // drain this wave's async copies before the barrier
    __syncthreads();

    // ---- compute: wave -> 16 output cols, two 16-row M-tiles sharing one B frag ----
    const int wave = tid >> 5;
    const int lane = tid & 31;
    const int half = lane >> 4;
    const int mn   = lane & 15;
    const int col  = wave * 16 + mn;

    v8f acc0 = {}, acc1 = {};
    // A k order used here (contiguous, same as B): lane(half,m): elem e -> K = k0 + 16*half + e
    const unsigned short* a0p = &sfeat[mn * LDS_STRIDE + 16 * half];
    const unsigned short* a1p = a0p + 16 * LDS_STRIDE;
    // B layout: lane(half,n): elem e -> B[K = k0 + 16*half + e][N=n] = W[col][k0 + 16*half + e]
    const unsigned short* wrh = Wh + (size_t)col * K_TOT + 16 * half;
    const float*          wrf = W  + (size_t)col * K_TOT + 16 * half;

#pragma unroll
    for (int k0 = 0; k0 < K_TOT; k0 += 32) {
        v16h bfrag;
        if (MODE >= 1) {
            bfrag = __builtin_bit_cast(v16h, *(const v16us*)(wrh + k0));
        } else {
            const v4f* wp = (const v4f*)(wrf + k0);
            v4f w0 = wp[0], w1 = wp[1], w2 = wp[2], w3 = wp[3];
            union { v16us u; unsigned int d[8]; } ub;
            ub.d[0] = pkh(w0.x, w0.y); ub.d[1] = pkh(w0.z, w0.w);
            ub.d[2] = pkh(w1.x, w1.y); ub.d[3] = pkh(w1.z, w1.w);
            ub.d[4] = pkh(w2.x, w2.y); ub.d[5] = pkh(w2.z, w2.w);
            ub.d[6] = pkh(w3.x, w3.y); ub.d[7] = pkh(w3.z, w3.w);
            bfrag = __builtin_bit_cast(v16h, ub.u);
        }

        union { v16us u; v8us h[2]; } ua;
        ua.h[0] = *(const v8us*)(a0p + k0);
        ua.h[1] = *(const v8us*)(a0p + k0 + 8);
        v16h af0 = __builtin_bit_cast(v16h, ua.u);
        acc0 = __builtin_amdgcn_wmma_f32_16x16x32_f16(false, af0, false, bfrag,
                                                      (short)0, acc0, false, false);

        ua.h[0] = *(const v8us*)(a1p + k0);
        ua.h[1] = *(const v8us*)(a1p + k0 + 8);
        v16h af1 = __builtin_bit_cast(v16h, ua.u);
        acc1 = __builtin_amdgcn_wmma_f32_16x16x32_f16(false, af1, false, bfrag,
                                                      (short)0, acc1, false, false);
    }

    // ---- epilogue: bias + NT stores (write-once stream; keep x resident in L2) ----
    const float bv = bias[col];
#pragma unroll
    for (int r = 0; r < 8; ++r) {
        const int row0 = edge0 + 8 * half + r;          // C/D layout: VGPR r -> M = r + 8*half
        const int row1 = row0 + 16;
        if (row0 < n_edges)
            __builtin_nontemporal_store(acc0[r] + bv, out + (size_t)row0 * N_OUT + col);
        if (row1 < n_edges)
            __builtin_nontemporal_store(acc1[r] + bv, out + (size_t)row1 * N_OUT + col);
    }
}

extern "C" void kernel_launch(void* const* d_in, const int* in_sizes, int n_in,
                              void* d_out, int out_size, void* d_ws, size_t ws_size,
                              hipStream_t stream) {
    const float* x  = (const float*)d_in[0];
    const int*   ei = (const int*)d_in[1];
    const float* ea = (const float*)d_in[2];
    const float* W  = (const float*)d_in[3];
    const float* b  = (const float*)d_in[4];
    float* out = (float*)d_out;

    const int n_edges = in_sizes[2] / N_IN_EDGE;            // edge_attr is [E, 64]
    const int nblk    = (n_edges + M_TILE - 1) / M_TILE;

    const int    wn      = N_OUT * K_TOT;                   // 40960 elements
    const int    xn      = in_sizes[0];                     // 50000*128 elements
    const size_t wh_b    = (size_t)wn * sizeof(unsigned short);
    const size_t xh_b    = (size_t)xn * sizeof(unsigned short);

    if (ws_size >= wh_b + xh_b) {
        unsigned short* Wh = (unsigned short*)d_ws;
        unsigned short* Xh = (unsigned short*)((char*)d_ws + wh_b);
        hipLaunchKernelGGL(cvt_f32_f16, dim3((wn / 8 + 255) / 256), dim3(256), 0, stream, W, Wh, wn);
        hipLaunchKernelGGL(cvt_f32_f16, dim3((xn / 8 + 255) / 256), dim3(256), 0, stream, x, Xh, xn);
        hipLaunchKernelGGL((edgeconv_wmma_f16<2>), dim3(nblk), dim3(256), 0, stream,
                           x, Xh, ei, ea, W, Wh, b, out, n_edges);
    } else if (ws_size >= wh_b) {
        unsigned short* Wh = (unsigned short*)d_ws;
        hipLaunchKernelGGL(cvt_f32_f16, dim3((wn / 8 + 255) / 256), dim3(256), 0, stream, W, Wh, wn);
        hipLaunchKernelGGL((edgeconv_wmma_f16<1>), dim3(nblk), dim3(256), 0, stream,
                           x, (const unsigned short*)nullptr, ei, ea, W, Wh, b, out, n_edges);
    } else {
        hipLaunchKernelGGL((edgeconv_wmma_f16<0>), dim3(nblk), dim3(256), 0, stream,
                           x, (const unsigned short*)nullptr, ei, ea, W,
                           (const unsigned short*)nullptr, b, out, n_edges);
    }
}
